// IsometryRegNoBackprop_67843303408034
// MI455X (gfx1250) — matrix operand order repaired and checked by reading, not compiled, not run on hardware
//
#include <hip/hip_runtime.h>
#include <math.h>

typedef __attribute__((ext_vector_type(8)))  float  v8f;
typedef __attribute__((ext_vector_type(16))) __bf16 v16bf;
typedef __attribute__((ext_vector_type(4)))  __bf16 v4bf;

#define WMMA_BF16(a, b, c) \
  __builtin_amdgcn_wmma_f32_16x16x32_bf16(false, (a), false, (b), (short)0, (c), false, false)

constexpr int B   = 512;
constexpr int D   = 3072;   // C*H*W
constexpr int HID = 4096;
constexpr int NC  = 10;
constexpr int M   = 9;      // NC-1
constexpr float EPS   = 0.1f;
constexpr float NSTAB = 1e-7f;

// A-fragment slot permutation for the 16x32 bf16 WMMA A layout:
// element i of a lane's v16bf holds K = 16*(i/8) + 8*hl + (i%8); storing
// K at slot ((K>>3)&1)*16 + (K>>4)*8 + (K&7) makes each lane's fragment one
// contiguous 32-byte run at offset hl*16.
__device__ __forceinline__ int aslot(int k) {
  return ((k >> 3) & 1) * 16 + ((k >> 4) << 3) + (k & 7);
}

// ---------------------------------------------------------------- init -----
__global__ void k_init(float* out, float* Gws) {
  int t = blockIdx.x * blockDim.x + threadIdx.x;
  if (t == 0) out[0] = 0.f;
  if (t < B * 81) Gws[t] = 0.f;
}

// ------------------------------------------- W1 -> Jacobian B operand ------
// B-fragment order over K=j: Bh/Bl[(j/32)*(D*32) + d*32 + aslot(j%32)].
__global__ void __launch_bounds__(256) k_splitB(const float* __restrict__ W1,
                                                __bf16* __restrict__ Bh,
                                                __bf16* __restrict__ Bl) {
  const size_t fidx = ((size_t)blockIdx.x * 256 + threadIdx.x) * 4;
  const int d = (int)(fidx >> 12);          // / HID
  const int j = (int)(fidx & (HID - 1));    // % HID
  float4 v = *(const float4*)(W1 + fidx);
  const size_t o = (size_t)(j >> 5) * ((size_t)D * 32) + (size_t)d * 32 + aslot(j & 31);
  v4bf hh, ll;
  float f[4] = {v.x, v.y, v.z, v.w};
#pragma unroll
  for (int i = 0; i < 4; ++i) {
    __bf16 h = (__bf16)f[i];
    hh[i] = h;
    ll[i] = (__bf16)(f[i] - (float)h);
  }
  *(v4bf*)(Bh + o) = hh;
  *(v4bf*)(Bl + o) = ll;
}

// ------------------------------------------- W1 -> forward B operand -------
// B-fragment order over K=d: Wfh/Wfl[(d/32)*(HID*32) + j*32 + aslot(d%32)].
// LDS-tiled so reads (along j) and writes (along d) are both coalesced.
__global__ void __launch_bounds__(256) k_splitWf(const float* __restrict__ W1,
                                                 __bf16* __restrict__ Wfh,
                                                 __bf16* __restrict__ Wfl) {
  __shared__ float tile[32][33];
  const int j0 = blockIdx.x * 32, d0 = blockIdx.y * 32;
  const int tx = threadIdx.x, ty = threadIdx.y;
#pragma unroll
  for (int i = 0; i < 4; ++i)
    tile[ty + i * 8][tx] = W1[(size_t)(d0 + ty + i * 8) * HID + j0 + tx];
  __syncthreads();
  const size_t base = (size_t)(d0 >> 5) * ((size_t)HID * 32);
#pragma unroll
  for (int i = 0; i < 4; ++i) {
    const float f = tile[tx][ty + i * 8];          // (d = d0+tx, j = j0+ty+i*8)
    const __bf16 h = (__bf16)f;
    const size_t o = base + (size_t)(j0 + ty + i * 8) * 32 + aslot(tx);
    Wfh[o] = h;
    Wfl[o] = (__bf16)(f - (float)h);
  }
}

// ------------------------------------------------ X -> forward A operand ---
// A-fragment-permuted order over K=d: Xph/Xpl[(d/32)*(B*32) + b*32 + aslot(d%32)].
__global__ void __launch_bounds__(256) k_splitX(const float* __restrict__ X,
                                                __bf16* __restrict__ Xph,
                                                __bf16* __restrict__ Xpl) {
  const int b = blockIdx.y;
  const int d = (blockIdx.x * 256 + threadIdx.x) * 4;     // D/4 = 768 per row
  float4 v = *(const float4*)(X + (size_t)b * D + d);
  const size_t o = (size_t)(d >> 5) * ((size_t)B * 32) + (size_t)b * 32 + aslot(d & 31);
  v4bf hh, ll;
  float f[4] = {v.x, v.y, v.z, v.w};
#pragma unroll
  for (int i = 0; i < 4; ++i) {                           // 4 consecutive slots
    __bf16 h = (__bf16)f[i];
    hh[i] = h;
    ll[i] = (__bf16)(f[i] - (float)h);
  }
  *(v4bf*)(Xph + o) = hh;
  *(v4bf*)(Xpl + o) = ll;
}

// ------------------------------------------------------------- forward -----
// H = relu(X @ W1 + b1), split-bf16, fragment-ordered operands: no LDS,
// no barriers; each wave owns one 16x32 output strip (2 n-tiles).
__global__ void __launch_bounds__(256) k_forward(const __bf16* __restrict__ Xph,
                                                 const __bf16* __restrict__ Xpl,
                                                 const __bf16* __restrict__ Wfh,
                                                 const __bf16* __restrict__ Wfl,
                                                 const float* __restrict__ bias,
                                                 float* __restrict__ H) {
  const int tid  = threadIdx.x;
  const int wave = tid >> 5, lane = tid & 31;
  const int hl = lane >> 4, ln = lane & 15;
  const int wm = wave >> 2, wn = wave & 3;       // 2 x 4 wave grid
  const int b0 = blockIdx.y * 32;
  const int j0 = blockIdx.x * 128;

  const size_t abase  = (size_t)(b0 + wm * 16 + ln) * 32 + hl * 16;
  const size_t bbase0 = (size_t)(j0 + wn * 32 + ln) * 32 + hl * 16;
  const size_t bbase1 = bbase0 + (size_t)16 * 32;

  v8f acc0 = {}, acc1 = {};
  for (int c = 0; c < D / 32; ++c) {
    const size_t ca = (size_t)c * ((size_t)B * 32) + abase;
    const size_t cb = (size_t)c * ((size_t)HID * 32);
    const v16bf ah  = *(const v16bf*)(Xph + ca);
    const v16bf al  = *(const v16bf*)(Xpl + ca);
    const v16bf bh0 = *(const v16bf*)(Wfh + cb + bbase0);
    const v16bf bl0 = *(const v16bf*)(Wfl + cb + bbase0);
    const v16bf bh1 = *(const v16bf*)(Wfh + cb + bbase1);
    const v16bf bl1 = *(const v16bf*)(Wfl + cb + bbase1);
    acc0 = WMMA_BF16(ah, bh0, acc0);
    acc0 = WMMA_BF16(ah, bl0, acc0);
    acc0 = WMMA_BF16(al, bh0, acc0);
    acc1 = WMMA_BF16(ah, bh1, acc1);
    acc1 = WMMA_BF16(ah, bl1, acc1);
    acc1 = WMMA_BF16(al, bh1, acc1);
  }
#pragma unroll
  for (int v = 0; v < 8; ++v) {
    const int row = b0 + wm * 16 + v + 8 * hl;
    const int ja  = j0 + wn * 32 + ln;
    const int jb  = ja + 16;
    float v0 = acc0[v] + bias[ja];
    float v1 = acc1[v] + bias[jb];
    H[(size_t)row * HID + ja] = v0 > 0.f ? v0 : 0.f;
    H[(size_t)row * HID + jb] = v1 > 0.f ? v1 : 0.f;
  }
}

// --------------------------------------------------- per-sample scalars ----
__global__ void __launch_bounds__(256) k_coeffs(const float* __restrict__ H,
                                                const float* __restrict__ W2,
                                                const float* __restrict__ b2,
                                                float* __restrict__ Aws,
                                                float* __restrict__ Fws) {
  __shared__ float ls[NC];
  const int b = blockIdx.x, tid = threadIdx.x;
  if (tid < NC) ls[tid] = b2[tid];
  __syncthreads();
  float acc[NC];
#pragma unroll
  for (int k = 0; k < NC; ++k) acc[k] = 0.f;
  for (int j = tid; j < HID; j += 256) {
    float hv = H[(size_t)b * HID + j];
    if (hv != 0.f) {
      const float* w = W2 + (size_t)j * NC;
#pragma unroll
      for (int k = 0; k < NC; ++k) acc[k] += hv * w[k];
    }
  }
#pragma unroll
  for (int k = 0; k < NC; ++k) atomicAdd(&ls[k], acc[k]);
  __syncthreads();
  if (tid == 0) {
    float p[NC], sp[NC];
    float mx = ls[0];
    for (int k = 1; k < NC; ++k) mx = fmaxf(mx, ls[k]);
    float s = 0.f;
    for (int k = 0; k < NC; ++k) { p[k] = expf(ls[k] - mx); s += p[k]; }
    const float inv = 1.f / s;
    for (int k = 0; k < NC; ++k) {
      p[k]  = p[k] * inv * (1.f - NC * NSTAB) + NSTAB;
      sp[k] = sqrtf(p[k]);
    }
    const float sp9 = sp[NC - 1], denom = 1.f - sp9;
    float dsum = 0.f;
    for (int k = 0; k < NC; ++k) dsum += sqrtf(p[k] * (1.f / NC));
    dsum = fminf(fmaxf(dsum, -1.f), 1.f);
    const float delta = 2.f * acosf(dsum);
    Fws[b] = delta * delta / (4.f * denom * denom * EPS * EPS);
    float* Ab = Aws + (size_t)b * 90;
    for (int m = 0; m < M; ++m) {
      const float dco = 1.f / (2.f * sp[m] * denom);
      const float lco = sp[m] / (2.f * sp9 * denom * denom);
      for (int c = 0; c < NC; ++c) {
        const float jm = p[m] * ((m == c ? 1.f : 0.f) - p[c]);
        const float j9 = p[NC - 1] * ((c == NC - 1 ? 1.f : 0.f) - p[c]);
        Ab[m * NC + c] = dco * jm + lco * j9;
      }
    }
  }
}

// ------------------------------------------------- per-sample big GEMM -----
// Jt[b] = V[b] @ W1^T via split-bf16 (Vh*Bh + Vh*Bl + Vl*Bh, f32 acc),
// double-buffered on-the-fly V, then G[b] += Jt Jt^T.
constexpr int SPLIT = 2;
constexpr int DSEG  = D / SPLIT;        // 1536 d-columns per block
constexpr int NWAVE = 16;               // 512 threads
constexpr int DW    = DSEG / NWAVE;     // 96 columns per wave
constexpr int NT    = DW / 16;          // 6 accumulator tiles per wave
constexpr int KT    = 32;               // one bf16 WMMA K-step
constexpr int NCHUNK = HID / KT;        // 128

__device__ __forceinline__ void compute_v(int j0, const float* As,
                                          const float* __restrict__ W2,
                                          const float* __restrict__ Hrow,
                                          __bf16 (*vh)[32], __bf16 (*vl)[32],
                                          int vrow, int vcol, int vslot) {
  const int j = j0 + vcol;
  const float* w = W2 + (size_t)j * NC;
  const float* a = As + vrow * NC;
  float u = 0.f;
#pragma unroll
  for (int k = 0; k < NC; ++k) u += a[k] * w[k];
  const float hv = Hrow[j];
  u = hv > 0.f ? u : 0.f;
  const __bf16 uh = (__bf16)u;
  vh[vrow][vslot] = uh;
  vl[vrow][vslot] = (__bf16)(u - (float)uh);
}

__global__ void __launch_bounds__(512) k_jacobian(const __bf16* __restrict__ Bh,
                                                  const __bf16* __restrict__ Bl,
                                                  const float* __restrict__ H,
                                                  const float* __restrict__ W2,
                                                  const float* __restrict__ Aws,
                                                  float* __restrict__ Gws) {
  __shared__ float As[16 * NC];
  __shared__ __align__(16) __bf16 Vsh[2][16][32];   // ping-pong A-frag hi
  __shared__ __align__(16) __bf16 Vsl[2][16][32];   // ping-pong A-frag lo
  __shared__ float Ts[NWAVE][16][17];
  const int b    = blockIdx.y;
  const int seg  = blockIdx.x;
  const int tid  = threadIdx.x;
  const int wave = tid >> 5, lane = tid & 31;
  const int hl = lane >> 4, ln = lane & 15;
  const int d0w   = seg * DSEG + wave * DW;
  const int vrow  = tid >> 5, vcol = tid & 31;
  const int vslot = aslot(vcol);
  const float* Hrow = H + (size_t)b * HID;

  if (tid < 16 * NC) As[tid] = (tid < M * NC) ? Aws[(size_t)b * 90 + tid] : 0.f;
  __syncthreads();
  compute_v(0, As, W2, Hrow, Vsh[0], Vsl[0], vrow, vcol, vslot);

  v8f acc[NT] = {};
  const size_t bbase = (size_t)(d0w + ln) * 32 + hl * 16;
  for (int c = 0; c < NCHUNK; ++c) {
    __syncthreads();                        // V[c] visible to all waves
    const int p = c & 1;
    const v16bf ah = *(const v16bf*)&Vsh[p][ln][hl * 16];
    const v16bf al = *(const v16bf*)&Vsl[p][ln][hl * 16];
    if (c + 1 < NCHUNK)                     // overlap next V with this chunk's WMMA
      compute_v((c + 1) * KT, As, W2, Hrow, Vsh[1 - p], Vsl[1 - p], vrow, vcol, vslot);
    const size_t co = (size_t)c * ((size_t)D * 32) + bbase;
    if (c + 1 < NCHUNK) {                   // gfx1250 prefetch of next B block
      __builtin_prefetch((const void*)(Bh + co + (size_t)D * 32), 0, 0);
      __builtin_prefetch((const void*)(Bl + co + (size_t)D * 32), 0, 0);
    }
#pragma unroll
    for (int t = 0; t < NT; ++t) {
      const v16bf bh = *(const v16bf*)(Bh + co + t * 512);
      const v16bf bl = *(const v16bf*)(Bl + co + t * 512);
      acc[t] = WMMA_BF16(ah, bh, acc[t]);
      acc[t] = WMMA_BF16(ah, bl, acc[t]);
      acc[t] = WMMA_BF16(al, bh, acc[t]);
    }
  }

  // partial G = Jt_tile @ Jt_tile^T per wave, then global accumulate
  float g0 = 0.f, g1 = 0.f, g2 = 0.f;
  const int e0 = lane, e1 = lane + 32, e2 = lane + 64;
#pragma unroll
  for (int t = 0; t < NT; ++t) {
    __syncthreads();
#pragma unroll
    for (int v = 0; v < 8; ++v) Ts[wave][v + 8 * hl][ln] = acc[t][v];
    __syncthreads();
    {
      const float(*T)[17] = Ts[wave];
      int m, n; float s;
      m = e0 / 9; n = e0 % 9; s = 0.f;
#pragma unroll
      for (int dd = 0; dd < 16; ++dd) s += T[m][dd] * T[n][dd];
      g0 += s;
      m = e1 / 9; n = e1 % 9; s = 0.f;
#pragma unroll
      for (int dd = 0; dd < 16; ++dd) s += T[m][dd] * T[n][dd];
      g1 += s;
      if (e2 < 81) {
        m = e2 / 9; n = e2 % 9; s = 0.f;
#pragma unroll
        for (int dd = 0; dd < 16; ++dd) s += T[m][dd] * T[n][dd];
        g2 += s;
      }
    }
  }
  atomicAdd(&Gws[(size_t)b * 81 + e0], g0);
  atomicAdd(&Gws[(size_t)b * 81 + e1], g1);
  if (e2 < 81) atomicAdd(&Gws[(size_t)b * 81 + e2], g2);
}

// ------------------------------------------------------------- reduce ------
__global__ void __launch_bounds__(128) k_reduce(const float* __restrict__ Gws,
                                                const float* __restrict__ Fws,
                                                float* __restrict__ out) {
  __shared__ float red[128];
  const int b = blockIdx.x, t = threadIdx.x;
  float s = 0.f;
  if (t < 81) {
    float d = Gws[(size_t)b * 81 + t];
    if (t % 10 == 0) d -= Fws[b];   // diagonal entries of 9x9
    s = d * d;
  }
  red[t] = s;
  __syncthreads();
  for (int off = 64; off > 0; off >>= 1) {
    if (t < off) red[t] += red[t + off];
    __syncthreads();
  }
  if (t == 0) atomicAdd(out, sqrtf(red[0]) * (1.f / (81.f * 512.f)));
}

// ------------------------------------------------------------- launch ------
extern "C" void kernel_launch(void* const* d_in, const int* in_sizes, int n_in,
                              void* d_out, int out_size, void* d_ws, size_t ws_size,
                              hipStream_t stream) {
  (void)in_sizes; (void)n_in; (void)out_size; (void)ws_size;
  const float* X  = (const float*)d_in[0];
  const float* W1 = (const float*)d_in[1];
  const float* b1 = (const float*)d_in[2];
  const float* W2 = (const float*)d_in[3];
  const float* b2 = (const float*)d_in[4];
  float* out = (float*)d_out;

  __bf16* Bh  = (__bf16*)d_ws;                      // HID*D   (25.2 MB)
  __bf16* Bl  = Bh  + (size_t)HID * D;              // HID*D   (25.2 MB)
  __bf16* Wfh = Bl  + (size_t)HID * D;              // D*HID   (25.2 MB)
  __bf16* Wfl = Wfh + (size_t)D * HID;              // D*HID   (25.2 MB)
  __bf16* Xph = Wfl + (size_t)D * HID;              // B*D     (3.1 MB)
  __bf16* Xpl = Xph + (size_t)B * D;                // B*D     (3.1 MB)
  float*  H   = (float*)(Xpl + (size_t)B * D);      // B*HID   (8 MB)
  float* Aws  = H + (size_t)B * HID;
  float* Fws  = Aws + (size_t)B * 90;
  float* Gws  = Fws + B;

  k_init<<<(B * 81 + 255) / 256, 256, 0, stream>>>(out, Gws);
  k_splitB<<<(D * HID / 4) / 256, 256, 0, stream>>>(W1, Bh, Bl);
  k_splitWf<<<dim3(HID / 32, D / 32), dim3(32, 8), 0, stream>>>(W1, Wfh, Wfl);
  k_splitX<<<dim3(D / 4 / 256, B), 256, 0, stream>>>(X, Xph, Xpl);
  k_forward<<<dim3(HID / 128, B / 32), 256, 0, stream>>>(Xph, Xpl, Wfh, Wfl, b1, H);
  k_coeffs<<<B, 256, 0, stream>>>(H, W2, b2, Aws, Fws);
  k_jacobian<<<dim3(SPLIT, B), 512, 0, stream>>>(Bh, Bl, H, W2, Aws, Gws);
  k_reduce<<<B, 128, 0, stream>>>(Gws, Fws, out);
}
